// MultiHeadAttention_47459388621430
// MI455X (gfx1250) — matrix operand order repaired and checked by reading, not compiled, not run on hardware
//
#include <hip/hip_runtime.h>
#include <hip/hip_bf16.h>

typedef _Float16 h16;
typedef __attribute__((ext_vector_type(16))) _Float16 v16h;
typedef __attribute__((ext_vector_type(8)))  _Float16 v8h;
typedef __attribute__((ext_vector_type(8)))  float    v8f;
typedef __attribute__((ext_vector_type(4)))  float    v4f;
typedef __attribute__((ext_vector_type(4)))  int      v4i;

constexpr int Sc  = 2048;
constexpr int Bc  = 2;
constexpr int Dm  = 1024;
constexpr int Hc  = 16;
constexpr int DKc = 64;
constexpr float SCALE = 0.125f; // 1/sqrt(64)

// ---- CDNA5 async global->LDS path (guarded; falls back to direct global) ----
#if defined(__has_builtin)
#if __has_builtin(__builtin_amdgcn_global_load_async_to_lds_b128) && \
    __has_builtin(__builtin_amdgcn_s_wait_asynccnt)
#define HAVE_ASYNC_LDS 1
#endif
#endif
#ifndef HAVE_ASYNC_LDS
#define HAVE_ASYNC_LDS 0
#endif

#if HAVE_ASYNC_LDS
typedef __attribute__((address_space(1))) v4i gv4i;   // global <4 x i32>
typedef __attribute__((address_space(3))) v4i lv4i;   // LDS    <4 x i32>
#endif

// ---------- WMMA fragment helpers (contiguous k = 16h + t order, same for A and B) ----------
// A 16x32 f16: lane L (m=L&15, h=L>>4): elems t = row[k0+16h+t]
__device__ inline v16h frag_a_f16(const h16* row, int k0, int half) {
  const h16* p0 = row + k0 + 16 * half;
  v8h x0 = *(const v8h*)p0;
  v8h x1 = *(const v8h*)(p0 + 8);
  v16h a;
#pragma unroll
  for (int t = 0; t < 8; ++t) { a[t] = x0[t]; a[8 + t] = x1[t]; }
  return a;
}

// B 32x16 f16 (as B^T rows, n-major): lane L (n=L&15, h=L>>4): elems t = rown[k0+16h+t]
__device__ inline v16h frag_b_f16(const h16* rown, int k0, int half) {
  const h16* p = rown + k0 + 16 * half;
  v8h x0 = *(const v8h*)p;
  v8h x1 = *(const v8h*)(p + 8);
  v16h b;
#pragma unroll
  for (int t = 0; t < 8; ++t) { b[t] = x0[t]; b[8 + t] = x1[t]; }
  return b;
}

__device__ inline v8f wmma32(v16h a, v16h b, v8f c) {
  return __builtin_amdgcn_wmma_f32_16x16x32_f16(false, a, false, b, (short)0, c, false, false);
}

// ---------- Kernel 0: f32 -> f16 conversion (one pass, vectorized x4) ----------
__global__ __launch_bounds__(256) void cvt_f32_f16_kernel(const float* __restrict__ src,
                                                          h16* __restrict__ dst, int n4) {
  const int i = blockIdx.x * 256 + threadIdx.x;
  if (i < n4) {
    v4f x = *(const v4f*)(src + (size_t)i * 4);
    h16* d = dst + (size_t)i * 4;
    d[0] = (h16)x[0]; d[1] = (h16)x[1]; d[2] = (h16)x[2]; d[3] = (h16)x[3];
  }
}

// ---------- Kernel 1: fused QKV projection (f16 GEMM, 32x64 tile per wave) ----------
// grid: x = (S*B)/32 row tiles, y = D/64 (== head), z = 0/1/2 for q/k/v
__global__ __launch_bounds__(32) void qkv_proj_kernel(
    const h16* __restrict__ Xqh, const h16* __restrict__ Xkh, const h16* __restrict__ Xvh,
    const h16* __restrict__ Winh, const float* __restrict__ bin,
    h16* __restrict__ Qh, h16* __restrict__ Kh, h16* __restrict__ Vt) {
  const int lane = threadIdx.x;
  const int half = lane >> 4, l16 = lane & 15;
  const int mt = blockIdx.x, nt = blockIdx.y, z = blockIdx.z;
  const h16* X = (z == 0) ? Xqh : (z == 1) ? Xkh : Xvh;
  const int row0 = mt * 32;
  const int n0 = nt * 64;
  const h16* a0row = X + (size_t)(row0 + l16) * Dm;
  const h16* a1row = X + (size_t)(row0 + 16 + l16) * Dm;
  const h16* wb = Winh + ((size_t)z * Dm + n0) * Dm;

  v8f acc[2][4];
#pragma unroll
  for (int g = 0; g < 2; ++g)
#pragma unroll
    for (int s = 0; s < 4; ++s)
#pragma unroll
      for (int r = 0; r < 8; ++r) acc[g][s][r] = 0.0f;

  for (int k0 = 0; k0 < Dm; k0 += 32) {
    v16h a0 = frag_a_f16(a0row, k0, half);
    v16h a1 = frag_a_f16(a1row, k0, half);
#pragma unroll
    for (int s = 0; s < 4; ++s) {
      v16h b = frag_b_f16(wb + (size_t)(s * 16 + l16) * Dm, k0, half);
      acc[0][s] = wmma32(a0, b, acc[0][s]);
      acc[1][s] = wmma32(a1, b, acc[1][s]);
    }
  }

#pragma unroll
  for (int s = 0; s < 4; ++s) {
    const int col = n0 + s * 16 + l16;      // C-frag: N = lane&15
    const float bias = bin[z * Dm + col];
    const int hh = nt, dk = s * 16 + l16;   // 64 cols per tile == one head
#pragma unroll
    for (int g = 0; g < 2; ++g) {
#pragma unroll
      for (int r = 0; r < 8; ++r) {
        const int M = g * 16 + r + 8 * half;
        const int rg = row0 + M;
        const int i = rg >> 1, bb = rg & 1;   // (S,B,D) row = i*B + b
        const float v = acc[g][s][r] + bias;
        if (z == 0)      Qh[(((size_t)(bb * Hc + hh)) * Sc + i) * DKc + dk] = (h16)v;
        else if (z == 1) Kh[(((size_t)(bb * Hc + hh)) * Sc + i) * DKc + dk] = (h16)v;
        else             Vt[(((size_t)(bb * Hc + hh)) * DKc + dk) * Sc + i] = (h16)v; // V^T
      }
    }
  }
}

// ---------- Kernel 2: flash attention (online softmax, async-LDS K staging) ----------
// grid: x = S/16 query tiles, y = H, z = B; one wave per block
__global__ __launch_bounds__(32) void flash_attn_kernel(
    const h16* __restrict__ Qh, const h16* __restrict__ Kh, const h16* __restrict__ Vt,
    const int* __restrict__ mask, h16* __restrict__ AO) {
  __shared__ h16 Ps[16 * 32];
#if HAVE_ASYNC_LDS
  __shared__ h16 Kt[2][32 * DKc];   // double-buffered 32x64 K tile (8 KB)
#endif
  const int lane = threadIdx.x;
  const int half = lane >> 4, l16 = lane & 15;
  const int i0 = blockIdx.x * 16;
  const int hh = blockIdx.y, bb = blockIdx.z;

  const h16* Qb = Qh + ((size_t)(bb * Hc + hh) * Sc) * DKc;
  const h16* Kb = Kh + ((size_t)(bb * Hc + hh) * Sc) * DKc;
  const h16* Vb = Vt + ((size_t)(bb * Hc + hh) * DKc) * Sc;
  const int* mb = mask + (size_t)bb * Sc * Sc;

  const h16* qrow = Qb + (size_t)(i0 + l16) * DKc;
  const v16h aq0 = frag_a_f16(qrow, 0, half);
  const v16h aq1 = frag_a_f16(qrow, 32, half);

#if HAVE_ASYNC_LDS
  // stage a 32-row x 64-half K tile: 256 16B segments, 8 per lane, ASYNCcnt-tracked
  auto stage_k = [&](int j0, int buf) {
#pragma unroll
    for (int t = 0; t < 8; ++t) {
      const int seg = t * 32 + lane;                // 0..255
      const int row = seg >> 3, c = (seg & 7) * 8;  // row 0..31, col chunk
      const h16* gp = Kb + (size_t)(j0 + row) * DKc + c;
      h16* lp = &Kt[buf][row * DKc + c];
      __builtin_amdgcn_global_load_async_to_lds_b128(
          (gv4i*)(void*)gp, (lv4i*)(void*)lp, 0, 0);
    }
  };
  stage_k(0, 0);
#endif

  v8f o[4];
  float rmax[8], rsum[8];
#pragma unroll
  for (int s = 0; s < 4; ++s)
#pragma unroll
    for (int r = 0; r < 8; ++r) o[s][r] = 0.0f;
#pragma unroll
  for (int r = 0; r < 8; ++r) { rmax[r] = -3.0e38f; rsum[r] = 0.0f; }

  for (int j0 = 0; j0 < Sc; j0 += 32) {
#if HAVE_ASYNC_LDS
    const int buf = (j0 >> 5) & 1;
    if (j0 + 32 < Sc) {
      stage_k(j0 + 32, buf ^ 1);                       // prefetch next tile
      __builtin_amdgcn_s_wait_asynccnt(8);             // current tile's 8 ops done
    } else {
      __builtin_amdgcn_s_wait_asynccnt(0);
    }
#else
    if (j0 + 32 < Sc)  // prefetch next K rows -> global_prefetch_b8
      __builtin_prefetch(Kb + (size_t)(j0 + 32 + l16) * DKc, 0, 0);
#endif

    // two 16x16 score tiles over K-dim = DK = 64 (2 WMMAs each)
    v8f s0, s1;
#pragma unroll
    for (int r = 0; r < 8; ++r) { s0[r] = 0.0f; s1[r] = 0.0f; }
    {
#if HAVE_ASYNC_LDS
      const h16* krow = &Kt[buf][(size_t)l16 * DKc];
#else
      const h16* krow = Kb + (size_t)(j0 + l16) * DKc;
#endif
      s0 = wmma32(aq0, frag_b_f16(krow, 0, half), s0);
      s0 = wmma32(aq1, frag_b_f16(krow, 32, half), s0);
    }
    {
#if HAVE_ASYNC_LDS
      const h16* krow = &Kt[buf][(size_t)(16 + l16) * DKc];
#else
      const h16* krow = Kb + (size_t)(j0 + 16 + l16) * DKc;
#endif
      s1 = wmma32(aq0, frag_b_f16(krow, 0, half), s1);
      s1 = wmma32(aq1, frag_b_f16(krow, 32, half), s1);
    }

    // scale + mask + online softmax, per C-fragment row (M = r + 8*half, N = l16)
#pragma unroll
    for (int r = 0; r < 8; ++r) {
      const int M = r + 8 * half;
      const int* mrow = mb + (size_t)(i0 + M) * Sc + j0;
      float x0 = s0[r] * SCALE; if (mrow[l16] == 1)      x0 = -1.0e9f;
      float x1 = s1[r] * SCALE; if (mrow[16 + l16] == 1) x1 = -1.0e9f;
      float mx = fmaxf(x0, x1);
      mx = fmaxf(mx, __shfl_xor(mx, 1));
      mx = fmaxf(mx, __shfl_xor(mx, 2));
      mx = fmaxf(mx, __shfl_xor(mx, 4));
      mx = fmaxf(mx, __shfl_xor(mx, 8));   // row reduce within 16-lane half
      const float nm = fmaxf(rmax[r], mx);
      const float p0 = __expf(x0 - nm);
      const float p1 = __expf(x1 - nm);
      const float al = __expf(rmax[r] - nm);
      rmax[r] = nm;
      float ps = p0 + p1;
      ps += __shfl_xor(ps, 1);
      ps += __shfl_xor(ps, 2);
      ps += __shfl_xor(ps, 4);
      ps += __shfl_xor(ps, 8);
      rsum[r] = rsum[r] * al + ps;
#pragma unroll
      for (int s = 0; s < 4; ++s) o[s][r] *= al;
      Ps[M * 32 + l16]      = (h16)p0;     // C-layout -> LDS
      Ps[M * 32 + 16 + l16] = (h16)p1;
    }
    __syncthreads();
    // re-read as A-fragment layout (16x32 f16)
    const v16h pa = frag_a_f16(&Ps[l16 * 32], 0, half);
    __syncthreads();

    // O += P @ V  (V stored transposed: contiguous in j)
#pragma unroll
    for (int ntl = 0; ntl < 4; ++ntl) {
      const h16* vrow = Vb + (size_t)(ntl * 16 + l16) * Sc;
      o[ntl] = wmma32(pa, frag_b_f16(vrow, j0, half), o[ntl]);
    }
  }

  // normalize and store attention output as f16 rows (i*B+b, D)
#pragma unroll
  for (int ntl = 0; ntl < 4; ++ntl) {
#pragma unroll
    for (int r = 0; r < 8; ++r) {
      const int M = r + 8 * half;
      const float v = o[ntl][r] / rsum[r];
      AO[(size_t)((i0 + M) * Bc + bb) * Dm + hh * 64 + ntl * 16 + l16] = (h16)v;
    }
  }
}

// ---------- Kernel 3: output projection (f16 GEMM, 32x64 tile per wave) ----------
__global__ __launch_bounds__(32) void out_proj_kernel(
    const h16* __restrict__ AO, const h16* __restrict__ Wouth,
    const float* __restrict__ bout, float* __restrict__ Out) {
  const int lane = threadIdx.x;
  const int half = lane >> 4, l16 = lane & 15;
  const int row0 = blockIdx.x * 32;
  const int n0 = blockIdx.y * 64;
  const h16* a0row = AO + (size_t)(row0 + l16) * Dm;
  const h16* a1row = AO + (size_t)(row0 + 16 + l16) * Dm;

  v8f acc[2][4];
#pragma unroll
  for (int g = 0; g < 2; ++g)
#pragma unroll
    for (int s = 0; s < 4; ++s)
#pragma unroll
      for (int r = 0; r < 8; ++r) acc[g][s][r] = 0.0f;

  for (int k0 = 0; k0 < Dm; k0 += 32) {
    v16h a0 = frag_a_f16(a0row, k0, half);
    v16h a1 = frag_a_f16(a1row, k0, half);
#pragma unroll
    for (int s = 0; s < 4; ++s) {
      v16h b = frag_b_f16(Wouth + (size_t)(n0 + s * 16 + l16) * Dm, k0, half);
      acc[0][s] = wmma32(a0, b, acc[0][s]);
      acc[1][s] = wmma32(a1, b, acc[1][s]);
    }
  }

#pragma unroll
  for (int s = 0; s < 4; ++s) {
    const int col = n0 + s * 16 + l16;
    const float bias = bout[col];
#pragma unroll
    for (int g = 0; g < 2; ++g) {
#pragma unroll
      for (int r = 0; r < 8; ++r) {
        const int M = g * 16 + r + 8 * half;
        Out[(size_t)(row0 + M) * Dm + col] = acc[g][s][r] + bias;
      }
    }
  }
}

extern "C" void kernel_launch(void* const* d_in, const int* in_sizes, int n_in,
                              void* d_out, int out_size, void* d_ws, size_t ws_size,
                              hipStream_t stream) {
  const float* Xq   = (const float*)d_in[0];
  const float* Xk   = (const float*)d_in[1];
  const float* Xv   = (const float*)d_in[2];
  const int*   mask = (const int*)d_in[3];
  const float* Win  = (const float*)d_in[4];
  const float* bin  = (const float*)d_in[5];
  const float* Wout = (const float*)d_in[6];
  const float* bout = (const float*)d_in[7];
  float* Out = (float*)d_out;

  const size_t NE = (size_t)Bc * Hc * Sc * DKc;   // 4M elements (== S*B*D)
  h16* ws = (h16*)d_ws;
  h16* Qh   = ws;
  h16* Kh   = ws + NE;
  h16* Vt   = ws + 2 * NE;
  h16* AO   = ws + 3 * NE;
  h16* Xqh  = ws + 4 * NE;
  h16* Xkh  = ws + 5 * NE;
  h16* Xvh  = ws + 6 * NE;
  h16* Winh = ws + 7 * NE;                         // 3M elements
  h16* Wouth = Winh + (size_t)3 * Dm * Dm;         // 1M elements

  // one-shot f32 -> f16 conversions (strips all cvts out of GEMM hot loops)
  const int nX4 = (int)(NE / 4);
  cvt_f32_f16_kernel<<<(nX4 + 255) / 256, 256, 0, stream>>>(Xq, Xqh, nX4);
  cvt_f32_f16_kernel<<<(nX4 + 255) / 256, 256, 0, stream>>>(Xk, Xkh, nX4);
  cvt_f32_f16_kernel<<<(nX4 + 255) / 256, 256, 0, stream>>>(Xv, Xvh, nX4);
  const int nWin4 = 3 * Dm * Dm / 4, nWout4 = Dm * Dm / 4;
  cvt_f32_f16_kernel<<<(nWin4 + 255) / 256, 256, 0, stream>>>(Win, Winh, nWin4);
  cvt_f32_f16_kernel<<<(nWout4 + 255) / 256, 256, 0, stream>>>(Wout, Wouth, nWout4);

  dim3 g1((Sc * Bc) / 32, Dm / 64, 3);
  qkv_proj_kernel<<<g1, 32, 0, stream>>>(Xqh, Xkh, Xvh, Winh, bin, Qh, Kh, Vt);

  dim3 g2(Sc / 16, Hc, Bc);
  flash_attn_kernel<<<g2, 32, 0, stream>>>(Qh, Kh, Vt, mask, AO);

  dim3 g3((Sc * Bc) / 32, Dm / 64);
  out_proj_kernel<<<g3, 32, 0, stream>>>(AO, Wouth, bout, Out);
}
